// ConformerYMT3SelfAttention_66718021976295
// MI455X (gfx1250) — matrix operand order repaired and checked by reading, not compiled, not run on hardware
//
#include <hip/hip_runtime.h>
#include <stdint.h>

#define B_  4
#define T_  1024
#define D_  1024
#define H_  16
#define HS_ 64
#define L_  2047

typedef __bf16 bf16_t;
typedef __attribute__((ext_vector_type(16))) __bf16    v16bf;
typedef __attribute__((ext_vector_type(8)))  float     v8f;
typedef __attribute__((ext_vector_type(4)))  unsigned  uint4v;
typedef __attribute__((ext_vector_type(8)))  unsigned  uint8v;
typedef __attribute__((ext_vector_type(4)))  float     float4v;

union FragU { v16bf v; unsigned u[8]; uint4v q[2]; };

__device__ __forceinline__ v8f zero8() {
  v8f z = {0.f, 0.f, 0.f, 0.f, 0.f, 0.f, 0.f, 0.f};
  return z;
}

__device__ __forceinline__ v8f wmma_bf16(v16bf a, v16bf b, v8f c) {
  // D = A(16x32 bf16) * B(32x16 bf16) + C(16x16 f32)
  return __builtin_amdgcn_wmma_f32_16x16x32_bf16(false, a, false, b, (short)0, c,
                                                 false, false);
}

__device__ __forceinline__ unsigned pack_bf16(float a, float b) {
  union { bf16_t h[2]; unsigned u; } p;
  p.h[0] = (bf16_t)a;
  p.h[1] = (bf16_t)b;
  return p.u;
}

// ---------------------------------------------------------------------------
// Tensor Data Mover: 2D tile (32 cols x 64 rows, 2-byte elements) -> LDS.
// D# built per CDNA5 ISA 8.3/8.4; 2-group form (VADDR2/3 NULL) for 2D tensors.
// tensor_dim0/1 are set to the *remaining* K/rows so OOB rows zero-fill LDS.
// All operands uniform -> SGPR groups via "s" constraints.
// ---------------------------------------------------------------------------
__device__ __forceinline__ void tdm_load_tile_2d(unsigned lds_addr,
                                                 const bf16_t* gptr,
                                                 unsigned tdim0, unsigned tdim1,
                                                 unsigned stride_elems) {
  unsigned long long ga = (unsigned long long)(uintptr_t)gptr;
  uint4v g0;
  g0[0] = 1u;                                   // count=1, user descriptor
  g0[1] = lds_addr;                             // LDS byte address
  g0[2] = (unsigned)(ga & 0xffffffffu);         // global_addr[31:0]
  g0[3] = (unsigned)((ga >> 32) & 0x01ffffffu)  // global_addr[56:32]
          | (2u << 30);                         // type = 2 ("image")
  uint8v g1;
  g1[0] = (1u << 16);                           // wg_mask=0, data_size=2B
  g1[1] = (tdim0 & 0xffffu) << 16;              // abar=0 | tensor_dim0[15:0]
  g1[2] = (tdim0 >> 16) | ((tdim1 & 0xffffu) << 16);  // td0[31:16]|td1[15:0]
  g1[3] = (tdim1 >> 16) | (32u << 16);          // td1[31:16] | tile_dim0=32
  g1[4] = 64u;                                  // tile_dim1=64, tile_dim2=0
  g1[5] = stride_elems;                         // tensor_dim0_stride[31:0]
  g1[6] = 0u;                                   // stride[47:32] | d1_stride lo
  g1[7] = 0u;                                   // d1_stride hi
  asm volatile("tensor_load_to_lds %0, %1" : : "s"(g0), "s"(g1) : "memory");
}

// ---------------------------------------------------------------------------
// fp32 -> bf16 conversion (vectorized x4; all our sizes are multiples of 4)
// ---------------------------------------------------------------------------
__global__ void cvt_f32_to_bf16(const float* __restrict__ in,
                                bf16_t* __restrict__ out, int n4) {
  int i = blockIdx.x * blockDim.x + threadIdx.x;
  if (i < n4) {
    float4v x = ((const float4v*)in)[i];
    unsigned* o = (unsigned*)out;
    o[2 * i]     = pack_bf16(x[0], x[1]);
    o[2 * i + 1] = pack_bf16(x[2], x[3]);
  }
}

// ---------------------------------------------------------------------------
// WMMA fragment loaders (wave32; contiguous k = 16*(lane/16) + s in both A and B)
// ---------------------------------------------------------------------------
// A 16x32 bf16 fragment from LDS tile (row stride 32 bf16).
__device__ __forceinline__ v16bf lds_a_frag(const bf16_t* As, int mbase) {
  int lane = threadIdx.x & 31;
  int row  = mbase + (lane & 15);
  int kb   = (lane >> 4) * 16;
  const unsigned* p = (const unsigned*)(As + row * 32);
  FragU f;
#pragma unroll
  for (int j = 0; j < 4; ++j) {
    f.u[j]     = p[(kb >> 1) + j];
    f.u[4 + j] = p[(kb >> 1) + 4 + j];
  }
  return f.v;
}

// B 32x16 bf16 fragment from LDS tile holding W rows: Bt[n][k], stride 32.
__device__ __forceinline__ v16bf lds_b_frag(const bf16_t* Bt, int nbase) {
  int lane = threadIdx.x & 31;
  int row  = nbase + (lane & 15);
  int kg   = (lane >> 4) * 16;
  const unsigned* p = (const unsigned*)(Bt + row * 32 + kg);
  FragU f;
#pragma unroll
  for (int j = 0; j < 8; ++j) f.u[j] = p[j];
  return f.v;
}

// B fragment: 16 contiguous bf16 at a per-lane base
__device__ __forceinline__ v16bf bfrag16(const bf16_t* base) {
  FragU f;
  f.q[0] = ((const uint4v*)base)[0];
  f.q[1] = ((const uint4v*)base)[1];
  return f.v;
}

// A fragment for Q rows in global memory: rowbase -> (b, q, h, 0), kb = lg*16
__device__ __forceinline__ v16bf qfrag(const bf16_t* rowbase, int kb) {
  FragU f;
  f.q[0] = *(const uint4v*)(rowbase + kb);
  f.q[1] = *(const uint4v*)(rowbase + kb + 8);
  return f.v;
}

// ---------------------------------------------------------------------------
// Tiled GEMM:  C(M,N) = A(M,K) * W(N,K)^T + bias      (bf16 in, f32 acc)
// LDS tiles staged by the Tensor Data Mover, double-buffered so the DMA of
// tile k+1 overlaps the WMMAs on tile k.
// MODE 0: out0 = float C
// MODE 1: out0 = bf16  C
// MODE 2: out0 = bf16 (C + pos_bias_u[n]);  out1 = bf16 (C + pos_bias_v[n])
// MODE 3: out0 = bf16 C written transposed as V_t[(b*H+h)*HS+hs][T]
// ---------------------------------------------------------------------------
template <int MODE>
__global__ void __launch_bounds__(256)
gemm_xwT(const bf16_t* __restrict__ A, const bf16_t* __restrict__ W,
         const float* __restrict__ bias, void* __restrict__ out0,
         void* __restrict__ out1, const float* __restrict__ posu,
         const float* __restrict__ posv, int M, int N, int K) {
  __shared__ bf16_t As[2][64 * 32];
  __shared__ bf16_t Bs[2][64 * 32];

  const int tid = threadIdx.x;
  const int m0 = blockIdx.x * 64;
  const int n0 = blockIdx.y * 64;
  const int w  = tid >> 5;
  const int mw = (w & 3) * 16;
  const int nw = (w >> 2) * 32;

  v8f acc0 = zero8();
  v8f acc1 = zero8();

  // LDS byte addresses (low 32 bits of the shared-array address)
  const unsigned ldsA = (unsigned)(uintptr_t)&As[0][0];
  const unsigned ldsB = (unsigned)(uintptr_t)&Bs[0][0];

  if (w == 0) {  // prologue: TDM-stage k0 = 0 into buffer 0
    tdm_load_tile_2d(ldsA, A + (size_t)m0 * K, (unsigned)K, (unsigned)(M - m0),
                     (unsigned)K);
    tdm_load_tile_2d(ldsB, W + (size_t)n0 * K, (unsigned)K, (unsigned)(N - n0),
                     (unsigned)K);
  }

  int buf = 0;
  for (int k0 = 0; k0 < K; k0 += 32, buf ^= 1) {
    if (w == 0) __builtin_amdgcn_s_wait_tensorcnt(0);
    __syncthreads();  // current buffer visible to all waves

    if (w == 0 && (k0 + 32) < K) {  // DMA next tile while we compute
      int k1 = k0 + 32;
      tdm_load_tile_2d(ldsA + (buf ^ 1) * 4096u, A + (size_t)m0 * K + k1,
                       (unsigned)(K - k1), (unsigned)(M - m0), (unsigned)K);
      tdm_load_tile_2d(ldsB + (buf ^ 1) * 4096u, W + (size_t)n0 * K + k1,
                       (unsigned)(K - k1), (unsigned)(N - n0), (unsigned)K);
    }

    v16bf a  = lds_a_frag(As[buf], mw);
    v16bf b0 = lds_b_frag(Bs[buf], nw);
    v16bf b1 = lds_b_frag(Bs[buf], nw + 16);
    acc0 = wmma_bf16(a, b0, acc0);
    acc1 = wmma_bf16(a, b1, acc1);
    __syncthreads();  // done reading buf before it is overwritten
  }

  const int lane = tid & 31;
  const int lg   = lane >> 4;
  const int n16  = lane & 15;
#pragma unroll
  for (int r = 0; r < 8; ++r) {
    int gm = m0 + mw + r + 8 * lg;
    if (gm >= M) continue;
#pragma unroll
    for (int s = 0; s < 2; ++s) {
      int gn = n0 + nw + 16 * s + n16;
      float v = (s == 0 ? acc0[r] : acc1[r]) + (bias ? bias[gn] : 0.f);
      if (MODE == 0) {
        ((float*)out0)[(size_t)gm * N + gn] = v;
      } else if (MODE == 1) {
        ((bf16_t*)out0)[(size_t)gm * N + gn] = (bf16_t)v;
      } else if (MODE == 2) {
        ((bf16_t*)out0)[(size_t)gm * N + gn] = (bf16_t)(v + posu[gn]);
        ((bf16_t*)out1)[(size_t)gm * N + gn] = (bf16_t)(v + posv[gn]);
      } else {  // MODE 3: V transposed (b,h,hs,t)
        int bb = gm >> 10, t = gm & 1023;
        int hh = gn >> 6, hsi = gn & 63;
        ((bf16_t*)out0)[(((size_t)(bb * H_ + hh)) * HS_ + hsi) * T_ + t] =
            (bf16_t)v;
      }
    }
  }
}

// ---------------------------------------------------------------------------
// Fused attention: one block per (b, h, 16-row q-block).
//  AC = Q_u * K^T (WMMA), BD = Q_v * P^T with rel-shift gather through LDS,
//  two-pass softmax (writes probs to d_out), probs*V (WMMA) -> ctx (bf16).
// ---------------------------------------------------------------------------
__global__ void __launch_bounds__(256)
attn_fused(const bf16_t* __restrict__ qu, const bf16_t* __restrict__ qv,
           const bf16_t* __restrict__ kmat, const bf16_t* __restrict__ vt,
           const bf16_t* __restrict__ pmat, float* __restrict__ probs,
           bf16_t* __restrict__ ctx) {
  __shared__ float bdw[8][16 * 32];   // per-wave BD strip tiles (16 KB)
  __shared__ float red[8][16 * 64];   // PV cross-wave reduction (32 KB)
  __shared__ float wred[8][16];
  __shared__ float rowstat[16];
  __shared__ float rowinv[16];

  const int bid = blockIdx.x;
  const int qb = bid & 63;
  const int h  = (bid >> 6) & 15;
  const int b  = bid >> 10;
  const int q0 = qb * 16;

  const int tid  = threadIdx.x;
  const int w    = tid >> 5;
  const int lane = tid & 31;
  const int lg   = lane >> 4;
  const int n16  = lane & 15;
  const int kb   = lg * 16;  // A-fragment K base within halves

  float* pblk = probs + ((size_t)(b * H_ + h) * T_ + q0) * T_;

  // --- Q fragments (A layout), HS=64 -> 2 k-steps of 32 ---
  const bf16_t* qrow  = qu + ((size_t)(b * T_ + q0 + n16) * D_) + h * HS_;
  const bf16_t* qvrow = qv + ((size_t)(b * T_ + q0 + n16) * D_) + h * HS_;
  v16bf qu0 = qfrag(qrow, kb),  qu1 = qfrag(qrow + 32, kb);
  v16bf qv0 = qfrag(qvrow, kb), qv1 = qfrag(qvrow + 32, kb);

  // ---------------- Pass A: scores = (AC + shifted BD) / 8 ----------------
  float pmax[8];
#pragma unroll
  for (int r = 0; r < 8; ++r) pmax[r] = -1e30f;

  for (int i = 0; i < 8; ++i) {
    int j0 = (w + i * 8) * 16;
    // AC: B-frag = K rows (j0+n16), contraction over hs
    v8f ac = zero8();
    {
      const bf16_t* kbase =
          kmat + ((size_t)(b * T_ + j0 + n16) * D_) + h * HS_ + lg * 16;
      ac = wmma_bf16(qu0, bfrag16(kbase), ac);
      ac = wmma_bf16(qu1, bfrag16(kbase + 32), ac);
    }
    // BD: two 16-wide strip tiles of Q_v * P^T covering c = j+1023-q
    int cb = j0 - q0 + 1008;  // always >= 0, multiple of 16
    v8f bd[2] = {zero8(), zero8()};
#pragma unroll
    for (int s = 0; s < 2; ++s) {
      int c = cb + 16 * s + n16;
      bool ok = (c < L_);
      int cc = ok ? c : 0;
      const bf16_t* pb = pmat + ((size_t)cc * H_ + h) * HS_ + lg * 16;
      uint4v z = {0u, 0u, 0u, 0u};
      FragU f0, f1;
      f0.q[0] = ok ? ((const uint4v*)pb)[0] : z;
      f0.q[1] = ok ? ((const uint4v*)pb)[1] : z;
      f1.q[0] = ok ? ((const uint4v*)(pb + 32))[0] : z;
      f1.q[1] = ok ? ((const uint4v*)(pb + 32))[1] : z;
      bd[s] = wmma_bf16(qv0, f0.v, bd[s]);
      bd[s] = wmma_bf16(qv1, f1.v, bd[s]);
    }
    // stash BD tiles (per-wave LDS; in-wave DS ordering guarantees RAW)
#pragma unroll
    for (int r = 0; r < 8; ++r) {
      int m = r + 8 * lg;
      bdw[w][m * 32 + n16]      = bd[0][r];
      bdw[w][m * 32 + 16 + n16] = bd[1][r];
    }
    // gather shifted BD (local col = n16 + 15 - m, in [0,30]) and emit raw
#pragma unroll
    for (int r = 0; r < 8; ++r) {
      int m = r + 8 * lg;
      float sc = (ac[r] + bdw[w][m * 32 + (n16 + 15 - m)]) * 0.125f;
      pmax[r] = fmaxf(pmax[r], sc);
      pblk[(size_t)m * T_ + j0 + n16] = sc;
    }
  }

  // row max: reduce over the 16 lanes of each half, then across waves
#pragma unroll
  for (int r = 0; r < 8; ++r) {
    float pm = pmax[r];
#pragma unroll
    for (int off = 1; off < 16; off <<= 1)
      pm = fmaxf(pm, __shfl_xor(pm, off, 32));
    if (n16 == 0) wred[w][lg * 8 + r] = pm;
  }
  __syncthreads();
  if (tid < 16) {
    float m = wred[0][tid];
    for (int ww = 1; ww < 8; ++ww) m = fmaxf(m, wred[ww][tid]);
    rowstat[tid] = m;
  }
  __syncthreads();

  // ---------------- Pass B: exp + row sums ----------------
  float psum[8];
#pragma unroll
  for (int r = 0; r < 8; ++r) psum[r] = 0.f;
  for (int i = 0; i < 8; ++i) {
    int j0 = (w + i * 8) * 16;
#pragma unroll
    for (int r = 0; r < 8; ++r) {
      int m = r + 8 * lg;
      size_t idx = (size_t)m * T_ + j0 + n16;
      float e = __expf(pblk[idx] - rowstat[m]);
      psum[r] += e;
      pblk[idx] = e;
    }
  }
#pragma unroll
  for (int r = 0; r < 8; ++r) {
    float ps = psum[r];
#pragma unroll
    for (int off = 1; off < 16; off <<= 1) ps += __shfl_xor(ps, off, 32);
    if (n16 == 0) wred[w][lg * 8 + r] = ps;
  }
  __syncthreads();
  if (tid < 16) {
    float s = 0.f;
    for (int ww = 0; ww < 8; ++ww) s += wred[ww][tid];
    rowinv[tid] = 1.f / s;
  }
  __syncthreads();

  // ---------------- Pass C: normalize probs in place ----------------
  for (int i = 0; i < 8; ++i) {
    int j0 = (w + i * 8) * 16;
#pragma unroll
    for (int r = 0; r < 8; ++r) {
      int m = r + 8 * lg;
      pblk[(size_t)m * T_ + j0 + n16] *= rowinv[m];
    }
  }
  __syncthreads();

  // ---------------- Pass D: ctx = probs * V (WMMA over j) ----------------
  v8f cacc[4] = {zero8(), zero8(), zero8(), zero8()};
  const float* prow = pblk + (size_t)n16 * T_;  // A-frag row M = lane&15
  for (int s = 0; s < 4; ++s) {
    int jbase = (w + s * 8) * 32;
    FragU fa;
#pragma unroll
    for (int i2 = 0; i2 < 4; ++i2) {
      fa.u[i2] = pack_bf16(prow[jbase + kb + 2 * i2],
                           prow[jbase + kb + 2 * i2 + 1]);
      fa.u[4 + i2] = pack_bf16(prow[jbase + kb + 8 + 2 * i2],
                               prow[jbase + kb + 9 + 2 * i2]);
    }
#pragma unroll
    for (int nt = 0; nt < 4; ++nt) {
      const bf16_t* vb = vt +
          (((size_t)(b * H_ + h)) * HS_ + nt * 16 + n16) * T_ + jbase + lg * 16;
      cacc[nt] = wmma_bf16(fa.v, bfrag16(vb), cacc[nt]);
    }
  }
#pragma unroll
  for (int nt = 0; nt < 4; ++nt)
#pragma unroll
    for (int r = 0; r < 8; ++r)
      red[w][(r + 8 * lg) * 64 + nt * 16 + n16] = cacc[nt][r];
  __syncthreads();
  for (int e = tid; e < 1024; e += 256) {
    float s = 0.f;
    for (int ww = 0; ww < 8; ++ww) s += red[ww][e];
    int m = e >> 6, col = e & 63;
    ctx[((size_t)(b * T_ + q0 + m)) * D_ + h * HS_ + col] = (bf16_t)s;
  }
}

// ---------------------------------------------------------------------------
extern "C" void kernel_launch(void* const* d_in, const int* in_sizes, int n_in,
                              void* d_out, int out_size, void* d_ws,
                              size_t ws_size, hipStream_t stream) {
  (void)in_sizes; (void)n_in; (void)out_size; (void)ws_size;

  const float* hsrc = (const float*)d_in[0];
  const float* rpe  = (const float*)d_in[1];
  const float* Wq   = (const float*)d_in[2];
  const float* bq   = (const float*)d_in[3];
  const float* Wk   = (const float*)d_in[4];
  const float* bk   = (const float*)d_in[5];
  const float* Wv   = (const float*)d_in[6];
  const float* bv   = (const float*)d_in[7];
  const float* Wo   = (const float*)d_in[8];
  const float* bo   = (const float*)d_in[9];
  const float* Wpos = (const float*)d_in[10];
  const float* pbu  = (const float*)d_in[11];
  const float* pbv  = (const float*)d_in[12];

  char* ws = (char*)d_ws;
  size_t off = 0;
  auto alloc = [&](size_t bytes) {
    char* p = ws + off;
    off += (bytes + 255) & ~(size_t)255;
    return p;
  };
  const size_t ND = (size_t)B_ * T_ * D_;  // 4,194,304
  const size_t LD = (size_t)L_ * D_;       // 2,096,128
  const size_t DD = (size_t)D_ * D_;

  bf16_t* hs_bf   = (bf16_t*)alloc(ND * 2);
  bf16_t* pe_bf   = (bf16_t*)alloc(LD * 2);
  bf16_t* wq_bf   = (bf16_t*)alloc(DD * 2);
  bf16_t* wk_bf   = (bf16_t*)alloc(DD * 2);
  bf16_t* wv_bf   = (bf16_t*)alloc(DD * 2);
  bf16_t* wo_bf   = (bf16_t*)alloc(DD * 2);
  bf16_t* wpos_bf = (bf16_t*)alloc(DD * 2);
  bf16_t* p_bf    = (bf16_t*)alloc(LD * 2);
  bf16_t* qu_bf   = (bf16_t*)alloc(ND * 2);
  bf16_t* qv_bf   = (bf16_t*)alloc(ND * 2);
  bf16_t* k_bf    = (bf16_t*)alloc(ND * 2);
  bf16_t* vt_bf   = (bf16_t*)alloc(ND * 2);
  bf16_t* ctx_bf  = (bf16_t*)alloc(ND * 2);

  auto cvt = [&](const float* src, bf16_t* dst, size_t n) {
    int n4 = (int)(n / 4);
    cvt_f32_to_bf16<<<(n4 + 255) / 256, 256, 0, stream>>>(src, dst, n4);
  };
  cvt(hsrc, hs_bf, ND);
  cvt(rpe, pe_bf, LD);
  cvt(Wq, wq_bf, DD);
  cvt(Wk, wk_bf, DD);
  cvt(Wv, wv_bf, DD);
  cvt(Wo, wo_bf, DD);
  cvt(Wpos, wpos_bf, DD);

  dim3 blk(256);
  dim3 gX((B_ * T_ + 63) / 64, D_ / 64);
  dim3 gP((L_ + 63) / 64, D_ / 64);

  // Q (dual output with pos biases), K, V (transposed), P projections
  gemm_xwT<2><<<gX, blk, 0, stream>>>(hs_bf, wq_bf, bq, qu_bf, qv_bf, pbu, pbv,
                                      B_ * T_, D_, D_);
  gemm_xwT<1><<<gX, blk, 0, stream>>>(hs_bf, wk_bf, bk, k_bf, nullptr, nullptr,
                                      nullptr, B_ * T_, D_, D_);
  gemm_xwT<3><<<gX, blk, 0, stream>>>(hs_bf, wv_bf, bv, vt_bf, nullptr, nullptr,
                                      nullptr, B_ * T_, D_, D_);
  gemm_xwT<1><<<gP, blk, 0, stream>>>(pe_bf, wpos_bf, nullptr, p_bf, nullptr,
                                      nullptr, nullptr, L_, D_, D_);

  // Fused scores + softmax + probs*V
  float* probs = (float*)d_out + ND;
  attn_fused<<<B_ * H_ * (T_ / 16), blk, 0, stream>>>(qu_bf, qv_bf, k_bf,
                                                      vt_bf, p_bf, probs,
                                                      ctx_bf);

  // Output projection -> d_out[0 .. B*T*D)
  gemm_xwT<0><<<gX, blk, 0, stream>>>(ctx_bf, wo_bf, bo, d_out, nullptr,
                                      nullptr, nullptr, B_ * T_, D_, D_);
}
